// TemporalBlock_22488448761991
// MI455X (gfx1250) — matrix operand order repaired and checked by reading, not compiled, not run on hardware
//
#include <hip/hip_runtime.h>
#include <hip/hip_bf16.h>

typedef _Float16 v16h __attribute__((ext_vector_type(16)));
typedef _Float16 v8h  __attribute__((ext_vector_type(8)));
typedef float    v8f  __attribute__((ext_vector_type(8)));

#define BB   2
#define TT   16
#define T_   17
#define HH   24
#define WW2  24
#define HW   576
#define DD   128
#define NHD  4
#define DH   32
#define MLPD 512
#define NVIS (BB*TT*HW)      // 18432
#define NROW (BB*T_*HW)      // 19584

__device__ __forceinline__ float wred(float v) {
    #pragma unroll
    for (int m = 16; m > 0; m >>= 1) v += __shfl_xor(v, m, 32);
    return v;
}

// ---------------- weight pack: W (KxN f32 row-major) -> WMMA B-fragment order f16 ----
// dst[((nt*ktiles+kt)*32 + lane)*16 + e] = W[kt*32 + (lane>>4)*16 + e, nt*16 + (lane&15)]
__global__ void pack_w_kernel(const float* __restrict__ W, _Float16* __restrict__ dst,
                              int K, int N) {
    int tid = blockIdx.x * blockDim.x + threadIdx.x;
    if (tid >= K * N) return;
    int e    = tid & 15;
    int lane = (tid >> 4) & 31;
    int t2   = tid >> 9;
    int ktiles = K >> 5;
    int kt = t2 % ktiles, nt = t2 / ktiles;
    int k = (kt << 5) + ((lane >> 4) << 4) + e;
    int n = (nt << 4) + (lane & 15);
    dst[tid] = (_Float16)W[(size_t)k * N + n];
}

// ---------------- generic WMMA GEMM: C = act(A@Wp + bias) (+resid) ------------------
// A: MxK f16 row-major. Wp: packed. One wave per 16x16 tile, loop K in steps of 32.
__global__ void gemm_wmma_kernel(const _Float16* __restrict__ A,
                                 const _Float16* __restrict__ Bp,
                                 const float* __restrict__ bias,
                                 const float* __restrict__ resid,
                                 float* __restrict__ C32,
                                 _Float16* __restrict__ C16,
                                 int M, int N, int K, int doGelu) {
    const int ktiles = K >> 5;
    const int ntiles = N >> 4;
    const int tiles  = (M >> 4) * ntiles;
    int tile = blockIdx.x * (blockDim.x >> 5) + (threadIdx.x >> 5);
    if (tile >= tiles) return;
    int mt = tile / ntiles, nt = tile % ntiles;
    int l  = threadIdx.x & 31;
    int mrow = (mt << 4) + (l & 15);
    const _Float16* arow = A + (size_t)mrow * K + ((l >> 4) << 4);
    const _Float16* bptr = Bp + ((size_t)nt * ktiles << 9) + ((size_t)l << 4);
    v8f acc = {};
    for (int kt = 0; kt < ktiles; ++kt) {
        v8h a0 = *(const v8h*)(arow + kt * 32);
        v8h a1 = *(const v8h*)(arow + kt * 32 + 8);
        v16h b = *(const v16h*)(bptr + ((size_t)kt << 9));
        v16h a;
        #pragma unroll
        for (int i = 0; i < 8; ++i) { a[i] = a0[i]; a[i + 8] = a1[i]; }
        acc = __builtin_amdgcn_wmma_f32_16x16x32_f16(false, a, false, b,
                                                     (short)0, acc, false, false);
    }
    int ncol = (nt << 4) + (l & 15);
    float bv = bias ? bias[ncol] : 0.f;
    #pragma unroll
    for (int i = 0; i < 8; ++i) {
        int r = (mt << 4) + ((l >> 4) << 3) + i;
        float v = acc[i] + bv;
        if (doGelu) v = 0.5f * v * (1.f + erff(v * 0.70710678118654752f));
        size_t idx = (size_t)r * N + ncol;
        if (resid) v += resid[idx];
        if (C32) C32[idx] = v;
        if (C16) C16[idx] = (_Float16)v;
    }
}

// ---------------- LayerNorm over D=128 rows (one wave per row) ----------------------
__global__ void ln_rows_kernel(const float* __restrict__ in,
                               const float* __restrict__ w, const float* __restrict__ b,
                               _Float16* __restrict__ out16, float* __restrict__ out32,
                               int M) {
    int row = blockIdx.x * (blockDim.x >> 5) + (threadIdx.x >> 5);
    if (row >= M) return;
    int l = threadIdx.x & 31;
    const float* x = in + (size_t)row * DD;
    float v0 = x[l], v1 = x[l + 32], v2 = x[l + 64], v3 = x[l + 96];
    float mean = wred(v0 + v1 + v2 + v3) * (1.f / 128.f);
    float d0 = v0 - mean, d1 = v1 - mean, d2 = v2 - mean, d3 = v3 - mean;
    float var = wred(d0 * d0 + d1 * d1 + d2 * d2 + d3 * d3) * (1.f / 128.f);
    float rstd = rsqrtf(var + 1e-5f);
    float y0 = d0 * rstd * w[l]      + b[l];
    float y1 = d1 * rstd * w[l + 32] + b[l + 32];
    float y2 = d2 * rstd * w[l + 64] + b[l + 64];
    float y3 = d3 * rstd * w[l + 96] + b[l + 96];
    size_t o = (size_t)row * DD;
    if (out16) { out16[o+l] = (_Float16)y0; out16[o+l+32] = (_Float16)y1;
                 out16[o+l+64] = (_Float16)y2; out16[o+l+96] = (_Float16)y3; }
    if (out32) { out32[o+l] = y0; out32[o+l+32] = y1; out32[o+l+64] = y2; out32[o+l+96] = y3; }
}

// ---------------- gather visual rows: query[:,1:] -> (NVIS,128) ---------------------
__global__ void gather_vis_kernel(const float* __restrict__ q, float* __restrict__ vis) {
    int tid = blockIdx.x * blockDim.x + threadIdx.x;
    if (tid >= NVIS * DD) return;
    int n = tid >> 7, d = tid & 127;
    int b = n / (TT * HW), rem = n % (TT * HW);
    vis[tid] = q[(((size_t)b * T_ + 1 + rem / HW) * HW + rem % HW) * DD + d];
}

// ---------------- mean over HW of rows at t=0 (works for query or X layout) ---------
__global__ void cls_mean_kernel(const float* __restrict__ base, float* __restrict__ out) {
    int tid = blockIdx.x * blockDim.x + threadIdx.x;
    if (tid >= BB * DD) return;
    int b = tid >> 7, d = tid & 127;
    float s = 0.f;
    for (int hw = 0; hw < HW; ++hw)
        s += base[(((size_t)b * T_) * HW + hw) * DD + d];
    out[tid] = s * (1.f / (float)HW);
}

// ---------------- tiny projections: [LN(cls b0), LN(cls b1), LN(0)=ln_b] @ W + bias --
__global__ void proj_small_kernel(const float* __restrict__ Sln, const float* __restrict__ zrow,
                                  const float* __restrict__ W, const float* __restrict__ bias,
                                  float* __restrict__ out) {
    int tid = blockIdx.x * blockDim.x + threadIdx.x;
    if (tid >= 3 * DD) return;
    int r = tid >> 7, d = tid & 127;
    const float* s = (r < 2) ? (Sln + (size_t)r * DD) : zrow;
    float acc = bias[d];
    for (int k = 0; k < DD; ++k) acc += s[k] * W[(size_t)k * DD + d];
    out[tid] = acc;
}

// ---------------- spatial attention (one wave per (row, head)) ----------------------
__global__ void spatial_attn_kernel(const float* __restrict__ Qs, const float* __restrict__ Ks,
                                    const float* __restrict__ Vs,
                                    const float* __restrict__ KS, const float* __restrict__ VS,
                                    _Float16* __restrict__ out16) {
    int w = blockIdx.x * (blockDim.x >> 5) + (threadIdx.x >> 5);
    int n = w >> 2, h = w & 3;
    if (n >= NVIS) return;
    int l  = threadIdx.x & 31;
    int bt = n / HW, pos = n % HW;
    int hh = pos / WW2, ww = pos % WW2;
    int cb = n % BB;                     // faithful torch .repeat ordering
    int hd = h * DH + l;
    float qv = Qs[(size_t)n * DD + hd];
    const float scale = 0.17677669529663687f;  // 1/sqrt(32)
    float sarr[26];
    // slot 0 = CLS key, but masked by validity of neighbor pattern 0 (shifted mask, faithful)
    {
        float t = wred(qv * KS[(size_t)cb * DD + hd]) * scale;
        bool oob0 = (hh - 2 < 0) || (ww - 2 < 0);
        sarr[0] = oob0 ? -1e9f : t;
    }
    for (int p = 0; p < 25; ++p) {       // slot p+1 uses neighbor p, masked by neighbor p+1
        int y = hh + p / 5 - 2, x = ww + p % 5 - 2;
        bool inb = (y >= 0) && (y < HH) && (x >= 0) && (x < WW2);
        float kv = inb ? Ks[(((size_t)bt * HW) + y * WW2 + x) * DD + hd]
                       : KS[2 * DD + hd];   // K of padded zero row
        float t = wred(qv * kv) * scale;
        bool masked = false;
        if (p + 1 < 25) {
            int y2 = hh + (p + 1) / 5 - 2, x2 = ww + (p + 1) % 5 - 2;
            masked = !((y2 >= 0) && (y2 < HH) && (x2 >= 0) && (x2 < WW2));
        }
        sarr[p + 1] = masked ? -1e9f : t;
    }
    float mx = sarr[0];
    #pragma unroll
    for (int s = 1; s < 26; ++s) mx = fmaxf(mx, sarr[s]);
    float den = 0.f;
    #pragma unroll
    for (int s = 0; s < 26; ++s) den += __expf(sarr[s] - mx);
    float inv = 1.f / den;
    float o = __expf(sarr[0] - mx) * inv * VS[(size_t)cb * DD + hd];
    for (int p = 0; p < 25; ++p) {
        int y = hh + p / 5 - 2, x = ww + p % 5 - 2;
        bool inb = (y >= 0) && (y < HH) && (x >= 0) && (x < WW2);
        float vv = inb ? Vs[(((size_t)bt * HW) + y * WW2 + x) * DD + hd]
                       : VS[2 * DD + hd];
        o += __expf(sarr[p + 1] - mx) * inv * vv;
    }
    out16[(size_t)n * DD + hd] = (_Float16)o;
}

// ---------------- temporal attention + logits (one wave per (seq, head)) -------------
__global__ void temporal_attn_kernel(const float* __restrict__ Qt, const float* __restrict__ Kt,
                                     const float* __restrict__ Vt,
                                     const unsigned char* __restrict__ tmask,
                                     _Float16* __restrict__ out16,
                                     float* __restrict__ logits) {
    int w = blockIdx.x * (blockDim.x >> 5) + (threadIdx.x >> 5);
    int n = w >> 2, h = w & 3;
    if (n >= BB * HW) return;
    int l  = threadIdx.x & 31;
    int hd = h * DH + l;
    const float scale = 0.17677669529663687f;
    for (int tq = 0; tq < T_; ++tq) {
        float qv = Qt[((size_t)n * T_ + tq) * DD + hd];
        float s[T_];
        float mx = -1e30f;
        for (int tk = 0; tk < T_; ++tk) {
            float t = wred(qv * Kt[((size_t)n * T_ + tk) * DD + hd]) * scale;
            if (tmask[((size_t)n * T_ + tq) * T_ + tk]) t = -1e9f;
            s[tk] = t;
            mx = fmaxf(mx, t);
        }
        if (l < T_)
            logits[((((size_t)n * NHD + h) * T_) + tq) * T_ + l] = s[l];
        float den = 0.f;
        #pragma unroll
        for (int tk = 0; tk < T_; ++tk) den += __expf(s[tk] - mx);
        float inv = 1.f / den, o = 0.f;
        for (int tk = 0; tk < T_; ++tk)
            o += __expf(s[tk] - mx) * Vt[((size_t)n * T_ + tk) * DD + hd];
        out16[((size_t)n * T_ + tq) * DD + hd] = (_Float16)(o * inv);
    }
}

// ---------------- assemble X (B,T_,HW,D): t=0 from query, t>0 spatial result ---------
__global__ void assemble_x_kernel(const float* __restrict__ q, const float* __restrict__ SOut,
                                  float* __restrict__ X) {
    int tid = blockIdx.x * blockDim.x + threadIdx.x;
    if (tid >= NROW * DD) return;
    int row = tid >> 7, d = tid & 127;
    int b = row / (T_ * HW), rem = row % (T_ * HW);
    int t = rem / HW, hw = rem % HW;
    if (t == 0) X[tid] = q[tid];
    else        X[tid] = SOut[(((size_t)(b * TT + (t - 1)) * HW) + hw) * DD + d];
}

// X (B,T_,HW,D) -> XT (B*HW, T_, D)
__global__ void x_to_xt_kernel(const float* __restrict__ X, float* __restrict__ XT) {
    int tid = blockIdx.x * blockDim.x + threadIdx.x;
    if (tid >= NROW * DD) return;
    int r = tid >> 7, d = tid & 127;
    int n = r / T_, t = r % T_;
    int b = n / HW, hw = n % HW;
    XT[tid] = X[((((size_t)b * T_ + t) * HW) + hw) * DD + d];
}

// XTout (B*HW,T_,D) -> X (B,T_,HW,D)
__global__ void xt_to_x_kernel(const float* __restrict__ XTo, float* __restrict__ X) {
    int tid = blockIdx.x * blockDim.x + threadIdx.x;
    if (tid >= NROW * DD) return;
    int row = tid >> 7, d = tid & 127;
    int b = row / (T_ * HW), rem = row % (T_ * HW);
    int t = rem / HW, hw = rem % HW;
    X[tid] = XTo[(((size_t)(b * HW + hw) * T_) + t) * DD + d];
}

// broadcast CLS mean back over all spatial positions at t=0
__global__ void cls_bcast_kernel(const float* __restrict__ cls, float* __restrict__ X) {
    int tid = blockIdx.x * blockDim.x + threadIdx.x;
    if (tid >= BB * HW * DD) return;
    int d = tid & 127, t2 = tid >> 7;
    int b = t2 / HW, hw = t2 % HW;
    X[(((size_t)b * T_) * HW + hw) * DD + d] = cls[(size_t)b * DD + d];
}

extern "C" void kernel_launch(void* const* d_in, const int* in_sizes, int n_in,
                              void* d_out, int out_size, void* d_ws, size_t ws_size,
                              hipStream_t stream) {
    const float* query = (const float*)d_in[0];
    const unsigned char* tmask = (const unsigned char*)d_in[2];
    const float* ln_s_w = (const float*)d_in[3];
    const float* ln_s_b = (const float*)d_in[4];
    const float* sWq = (const float*)d_in[5];  const float* sbq = (const float*)d_in[6];
    const float* sWk = (const float*)d_in[7];  const float* sbk = (const float*)d_in[8];
    const float* sWv = (const float*)d_in[9];  const float* sbv = (const float*)d_in[10];
    const float* sWo = (const float*)d_in[11]; const float* sbo = (const float*)d_in[12];
    const float* ln_t_w = (const float*)d_in[13];
    const float* ln_t_b = (const float*)d_in[14];
    const float* tWq = (const float*)d_in[15]; const float* tbq = (const float*)d_in[16];
    const float* tWk = (const float*)d_in[17]; const float* tbk = (const float*)d_in[18];
    const float* tWv = (const float*)d_in[19]; const float* tbv = (const float*)d_in[20];
    const float* tWo = (const float*)d_in[21]; const float* tbo = (const float*)d_in[22];
    const float* ln_m_w = (const float*)d_in[23];
    const float* ln_m_b = (const float*)d_in[24];
    const float* mW1 = (const float*)d_in[25]; const float* mb1 = (const float*)d_in[26];
    const float* mW2 = (const float*)d_in[27]; const float* mb2 = (const float*)d_in[28];

    float* out_x      = (float*)d_out;
    float* out_logits = out_x + (size_t)BB * T_ * HW * DD;   // 2,506,752

    char* ws = (char*)d_ws;
    size_t off = 0;
    auto take = [&](size_t bytes) -> void* {
        void* p = ws + off;
        off += (bytes + 255) & ~(size_t)255;
        return p;
    };
    _Float16* wp   = (_Float16*)take(262144 * 2);
    float*    vis  = (float*)   take((size_t)NVIS * DD * 4);
    _Float16* ln16 = (_Float16*)take((size_t)NROW * DD * 2);
    _Float16* at16 = (_Float16*)take((size_t)NROW * DD * 2);
    float*    Qb   = (float*)   take((size_t)NROW * DD * 4);
    float*    Kb   = (float*)   take((size_t)NROW * DD * 4);
    float*    Vb   = (float*)   take((size_t)NROW * DD * 4);
    float*    SOut = (float*)   take((size_t)NVIS * DD * 4);
    float*    X    = (float*)   take((size_t)NROW * DD * 4);
    float*    XT   = (float*)   take((size_t)NROW * DD * 4);
    float*    XTo  = (float*)   take((size_t)NROW * DD * 4);
    _Float16* H16  = (_Float16*)take((size_t)NROW * MLPD * 2);
    float*    clsF = (float*)   take(BB * DD * 4);
    float*    Sln  = (float*)   take(BB * DD * 4);
    float*    KS   = (float*)   take(3 * DD * 4);
    float*    VS   = (float*)   take(3 * DD * 4);

    // packed-weight half offsets
    const size_t O_sWq = 0, O_sWk = 16384, O_sWv = 32768, O_sWo = 49152;
    const size_t O_tWq = 65536, O_tWk = 81920, O_tWv = 98304, O_tWo = 114688;
    const size_t O_m1 = 131072, O_m2 = 196608;

    auto pack = [&](const float* W, size_t dst, int K, int N) {
        int n = K * N;
        pack_w_kernel<<<(n + 255) / 256, 256, 0, stream>>>(W, wp + dst, K, N);
    };
    pack(sWq, O_sWq, DD, DD); pack(sWk, O_sWk, DD, DD);
    pack(sWv, O_sWv, DD, DD); pack(sWo, O_sWo, DD, DD);
    pack(tWq, O_tWq, DD, DD); pack(tWk, O_tWk, DD, DD);
    pack(tWv, O_tWv, DD, DD); pack(tWo, O_tWo, DD, DD);
    pack(mW1, O_m1, DD, MLPD); pack(mW2, O_m2, MLPD, DD);

    auto gemm = [&](const _Float16* A, size_t wOff, const float* bias, const float* resid,
                    float* C32, _Float16* C16, int M, int N, int K, int gelu) {
        int tiles = (M / 16) * (N / 16);
        gemm_wmma_kernel<<<(tiles + 3) / 4, 128, 0, stream>>>(A, wp + wOff, bias, resid,
                                                              C32, C16, M, N, K, gelu);
    };

    // ---- spatial branch ----
    gather_vis_kernel<<<(NVIS * DD + 255) / 256, 256, 0, stream>>>(query, vis);
    cls_mean_kernel<<<1, 256, 0, stream>>>(query, clsF);
    ln_rows_kernel<<<(NVIS + 3) / 4, 128, 0, stream>>>(vis, ln_s_w, ln_s_b, ln16, nullptr, NVIS);
    ln_rows_kernel<<<1, 128, 0, stream>>>(clsF, ln_s_w, ln_s_b, nullptr, Sln, BB);
    proj_small_kernel<<<3, 128, 0, stream>>>(Sln, ln_s_b, sWk, sbk, KS);
    proj_small_kernel<<<3, 128, 0, stream>>>(Sln, ln_s_b, sWv, sbv, VS);
    gemm(ln16, O_sWq, sbq, nullptr, Qb, nullptr, NVIS, DD, DD, 0);
    gemm(ln16, O_sWk, sbk, nullptr, Kb, nullptr, NVIS, DD, DD, 0);
    gemm(ln16, O_sWv, sbv, nullptr, Vb, nullptr, NVIS, DD, DD, 0);
    spatial_attn_kernel<<<NVIS, 128, 0, stream>>>(Qb, Kb, Vb, KS, VS, at16);
    gemm(at16, O_sWo, sbo, vis, SOut, nullptr, NVIS, DD, DD, 0);
    assemble_x_kernel<<<(NROW * DD + 255) / 256, 256, 0, stream>>>(query, SOut, X);

    // ---- temporal branch ----
    x_to_xt_kernel<<<(NROW * DD + 255) / 256, 256, 0, stream>>>(X, XT);
    ln_rows_kernel<<<(NROW + 3) / 4, 128, 0, stream>>>(XT, ln_t_w, ln_t_b, ln16, nullptr, NROW);
    gemm(ln16, O_tWq, tbq, nullptr, Qb, nullptr, NROW, DD, DD, 0);
    gemm(ln16, O_tWk, tbk, nullptr, Kb, nullptr, NROW, DD, DD, 0);
    gemm(ln16, O_tWv, tbv, nullptr, Vb, nullptr, NROW, DD, DD, 0);
    temporal_attn_kernel<<<(BB * HW * NHD + 3) / 4, 128, 0, stream>>>(Qb, Kb, Vb, tmask,
                                                                      at16, out_logits);
    gemm(at16, O_tWo, tbo, XT, XTo, nullptr, NROW, DD, DD, 0);
    xt_to_x_kernel<<<(NROW * DD + 255) / 256, 256, 0, stream>>>(XTo, X);
    cls_mean_kernel<<<1, 256, 0, stream>>>(X, clsF);
    cls_bcast_kernel<<<(BB * HW * DD + 255) / 256, 256, 0, stream>>>(clsF, X);

    // ---- MLP ----
    ln_rows_kernel<<<(NROW + 3) / 4, 128, 0, stream>>>(X, ln_m_w, ln_m_b, ln16, nullptr, NROW);
    gemm(ln16, O_m1, mb1, nullptr, nullptr, H16, NROW, MLPD, DD, 1);
    gemm(H16, O_m2, mb2, X, out_x, nullptr, NROW, DD, MLPD, 0);
}
